// PAKA3x3_84696755077659
// MI455X (gfx1250) — matrix operand order repaired and checked by reading, not compiled, not run on hardware
//
#include <hip/hip_runtime.h>

// ---------------------------------------------------------------------------
// PAKA 3x3 on gfx1250: bf16-WMMA pipeline with async global->LDS tile copies.
//   k_prep_w: repack all weights to bf16 WMMA-A layouts in workspace
//   k_prep_x: x fp32 [b][c][n] -> bf16 [b][n][c] (pixel-major, 256B rows)
//   k_convc : cc = wc2 * relu(wc1 * x)            -> ws (bf16, [b][pix][c])
//   k_convd : dd = wd2 * relu(conv3x3_reflect(x)) -> ws (bf16, [b][pix][16])
//   k_final : out = Wpaka * (patches .* (1 + tanh(dd_k + cc_c)))
// GEMMs: v_wmma_f32_16x16x32_bf16.  Tile fills: global_load_async_to_lds_b128.
// ---------------------------------------------------------------------------

#define BATCH 4
#define CCH   128
#define HH    128
#define WW    128
#define HWSZ  (HH * WW)
#define LDK   136   // padded K-stride (elems) for LDS bf16 tiles: 16B-aligned rows

typedef __attribute__((ext_vector_type(16))) __bf16 v16bf;
typedef __attribute__((ext_vector_type(8)))  float  v8f;

// ---- bf16 helpers (round-to-nearest-even) ---------------------------------
static __device__ __forceinline__ unsigned short f2bf(float f) {
  unsigned u = __float_as_uint(f);
  unsigned r = u + 0x7FFFu + ((u >> 16) & 1u);
  return (unsigned short)(r >> 16);
}
static __device__ __forceinline__ float bf2f(unsigned short h) {
  return __uint_as_float(((unsigned)h) << 16);
}
static __device__ __forceinline__ unsigned pack2(float a, float b) {
  return (unsigned)f2bf(a) | ((unsigned)f2bf(b) << 16);
}
static __device__ __forceinline__ uint4 pack8(v8f a) {
  uint4 r;
  r.x = pack2(a[0], a[1]); r.y = pack2(a[2], a[3]);
  r.z = pack2(a[4], a[5]); r.w = pack2(a[6], a[7]);
  return r;
}
static __device__ __forceinline__ uint4 pack8_relu(v8f a) {
  uint4 r;
  r.x = pack2(fmaxf(a[0], 0.f), fmaxf(a[1], 0.f));
  r.y = pack2(fmaxf(a[2], 0.f), fmaxf(a[3], 0.f));
  r.z = pack2(fmaxf(a[4], 0.f), fmaxf(a[5], 0.f));
  r.w = pack2(fmaxf(a[6], 0.f), fmaxf(a[7], 0.f));
  return r;
}

// ---- async global->LDS copy (16B per lane, ASYNCcnt-tracked) --------------
// The LDS destination is a VGPR byte address, so pass the wave-relative LDS byte
// offset = low 32 bits of the generic pointer (flat LDS addressing truncates).
static __device__ __forceinline__ void async_cp16(void* lds_dst, const void* gsrc) {
  unsigned ldso = (unsigned)(unsigned long long)lds_dst;
  asm volatile("global_load_async_to_lds_b128 %0, %1, off"
               :: "v"(ldso), "v"(gsrc) : "memory");
}
static __device__ __forceinline__ void async_wait0() {
  asm volatile("s_wait_asynccnt 0" ::: "memory");
}

// ---- WMMA fragment loaders (ISA 7.12.2 layouts, wave32) -------------------
// Source buffer: row r (M for A, N for B) has K contiguous, row stride ld.
// A 16x32 bf16: lane<16 -> v0..3=K0..7, v4..7=K16..23 ; lane>=16 -> K8..15,K24..31
static __device__ __forceinline__ v16bf load_a(const unsigned short* p, int ld) {
  const int lane = threadIdx.x & 31;
  const int r = lane & 15, hg = lane >> 4;
  const unsigned short* q = p + r * ld + hg * 8;
  union { uint4 u[2]; v16bf v; } t;
  t.u[0] = *(const uint4*)(q);
  t.u[1] = *(const uint4*)(q + 16);
  return t.v;
}
// B 32x16 bf16: lane<16 -> K0..7,K16..23 ; lane>=16 -> K8..15,K24..31 (matches A)
static __device__ __forceinline__ v16bf load_b(const unsigned short* p, int ld) {
  const int lane = threadIdx.x & 31;
  const int r = lane & 15, hg = lane >> 4;
  const unsigned short* q = p + r * ld + hg * 8;
  union { uint4 u[2]; v16bf v; } t;
  t.u[0] = *(const uint4*)(q);
  t.u[1] = *(const uint4*)(q + 16);
  return t.v;
}
static __device__ __forceinline__ v8f wmma_bf(v16bf a, v16bf b, v8f c) {
  return __builtin_amdgcn_wmma_f32_16x16x32_bf16(false, a, false, b, (short)0, c,
                                                 false, false);
}
static __device__ __forceinline__ v8f vzero8() {
  v8f z = {0.f, 0.f, 0.f, 0.f, 0.f, 0.f, 0.f, 0.f};
  return z;
}
static __device__ __forceinline__ int refl(int i, int n) {
  i = (i < 0) ? -i : i;
  return (i >= n) ? (2 * n - 2 - i) : i;
}

// ---------------------------------------------------------------------------
// Kernel 0a: weight repack fp32 -> bf16 WMMA-A layouts.
// ---------------------------------------------------------------------------
__global__ __launch_bounds__(256) void k_prep_w(
    const float* __restrict__ wc1, const float* __restrict__ wc2,
    const float* __restrict__ wd1, const float* __restrict__ wd2,
    const float* __restrict__ wpaka,
    unsigned short* __restrict__ wc1b, unsigned short* __restrict__ wc2b,
    unsigned short* __restrict__ wd1b, unsigned short* __restrict__ wd2b,
    unsigned short* __restrict__ wpb) {
  int i = blockIdx.x * 256 + threadIdx.x;
  if (i < 16384) { wc1b[i] = f2bf(wc1[i]); wc2b[i] = f2bf(wc2[i]); }
  if (i < 2048) {
    int m = i >> 7, ci = i & 127;
    wd2b[i] = (m < 9) ? f2bf(wd2[m * 128 + ci]) : (unsigned short)0;
  }
  if (i < 147456) {
    // wd1 flat: ((o*128+c)*3+kh)*3+kw = (o*128+c)*9 + kp -> wd1b[kp][o][c]
    int kp = i / 16384, oc = i - kp * 16384;
    wd1b[i] = f2bf(wd1[oc * 9 + kp]);
    // w_paka flat: (o*128+c)*9 + kp  ->  wpb[o*1152 + kp*128 + c]
    int o = i / 1152, t = i - o * 1152;
    int kq = t >> 7, c = t & 127;
    wpb[i] = f2bf(wpaka[(o * 128 + c) * 9 + kq]);
  }
}

// ---------------------------------------------------------------------------
// Kernel 0b: x fp32 [b][c][hw] -> bf16 [b][hw][c] (c contiguous). Reads hit L2
// (x = 32MB < 192MB L2); writes are coalesced.
// ---------------------------------------------------------------------------
__global__ __launch_bounds__(256) void k_prep_x(const float* __restrict__ x,
                                                unsigned short* __restrict__ xb16) {
  int i = blockIdx.x * 256 + threadIdx.x;   // [0, B*HW*C)
  int c = i & 127;
  int t = i >> 7;
  int n = t & (HWSZ - 1);
  int b = t >> 14;
  xb16[i] = f2bf(x[((size_t)(b * CCH + c) << 14) + n]);
}

// ---------------------------------------------------------------------------
// Kernel 1: cc = wc2 * relu(wc1 * x).  64-pixel tiles, 8 waves x 16 rows.
// ---------------------------------------------------------------------------
__global__ __launch_bounds__(256) void k_convc(
    const unsigned short* __restrict__ xb16,
    const unsigned short* __restrict__ wc1b,
    const unsigned short* __restrict__ wc2b, unsigned short* __restrict__ cc) {
  __shared__ unsigned short Xt[64 * LDK];  // [pix][ch]
  __shared__ unsigned short Tt[64 * LDK];  // relu(wc1*x), [pix][ch]
  const int tid = threadIdx.x;
  const int b = blockIdx.x >> 8;
  const int n0 = (blockIdx.x & 255) << 6;
  const unsigned short* xb = xb16 + ((size_t)(b * HWSZ + n0) << 7);

  // async fill: 64 rows x 256B, 16B per op
  for (int idx = tid; idx < 64 * 16; idx += 256) {
    int p = idx >> 4, c8 = (idx & 15) << 3;
    async_cp16(Xt + p * LDK + c8, xb + p * 128 + c8);
  }
  async_wait0();
  __syncthreads();

  const int wv = tid >> 5, m0 = wv << 4;
  const int lane = tid & 31, r = lane & 15, hg = lane >> 4;

  v16bf a[4];
#pragma unroll
  for (int kk = 0; kk < 4; kk++) a[kk] = load_a(wc1b + m0 * 128 + kk * 32, 128);
#pragma unroll
  for (int nt = 0; nt < 4; nt++) {
    v8f acc = vzero8();
#pragma unroll
    for (int kk = 0; kk < 4; kk++)
      acc = wmma_bf(a[kk], load_b(Xt + (nt * 16) * LDK + kk * 32, LDK), acc);
    int n = nt * 16 + r;
    *(uint4*)(Tt + n * LDK + m0 + hg * 8) = pack8_relu(acc);
  }
  __syncthreads();

#pragma unroll
  for (int kk = 0; kk < 4; kk++) a[kk] = load_a(wc2b + m0 * 128 + kk * 32, 128);
  unsigned short* ccb = cc + (((size_t)b * HWSZ + n0) << 7);
#pragma unroll
  for (int nt = 0; nt < 4; nt++) {
    v8f acc = vzero8();
#pragma unroll
    for (int kk = 0; kk < 4; kk++)
      acc = wmma_bf(a[kk], load_b(Tt + (nt * 16) * LDK + kk * 32, LDK), acc);
    int n = nt * 16 + r;
    *(uint4*)(ccb + n * 128 + m0 + hg * 8) = pack8(acc);
  }
}

// ---------------------------------------------------------------------------
// Kernel 2: dd = wd2 * relu(conv3x3_reflect(x, wd1)).  One row, 64-col tiles.
// ---------------------------------------------------------------------------
__global__ __launch_bounds__(256) void k_convd(
    const unsigned short* __restrict__ xb16,
    const unsigned short* __restrict__ wd1b,
    const unsigned short* __restrict__ wd2b, unsigned short* __restrict__ dd) {
  extern __shared__ unsigned short sm2[];
  unsigned short* Xh = sm2;                 // [3][66][LDK] halo tile (reflect)
  unsigned short* Tt = sm2 + 3 * 66 * LDK;  // [64][LDK]
  const int tid = threadIdx.x;
  const int b = blockIdx.x >> 8;
  const int rem = blockIdx.x & 255;
  const int h = rem >> 1, w0 = (rem & 1) << 6;
  const unsigned short* xb = xb16 + ((size_t)b * HWSZ << 7);

  // async fill: 3*66 rows x 256B (reflect => always in-bounds)
  for (int idx = tid; idx < 3 * 66 * 16; idx += 256) {
    int c8 = (idx & 15) << 3, t = idx >> 4;   // t in [0, 198)
    int col = t % 66, rr = t / 66;
    int wi = refl(w0 - 1 + col, WW);
    int hi = refl(h - 1 + rr, HH);
    async_cp16(Xh + (rr * 66 + col) * LDK + c8, xb + ((hi * WW + wi) << 7) + c8);
  }
  async_wait0();
  __syncthreads();

  const int wv = tid >> 5, m0 = wv << 4;
  const int lane = tid & 31, r = lane & 15, hg = lane >> 4;

  v8f acc[4];
#pragma unroll
  for (int nt = 0; nt < 4; nt++) acc[nt] = vzero8();

  for (int kp = 0; kp < 9; kp++) {
    int kh = kp / 3, kw = kp - kh * 3;
    v16bf a[4];
#pragma unroll
    for (int kk = 0; kk < 4; kk++)
      a[kk] = load_a(wd1b + kp * 16384 + m0 * 128 + kk * 32, 128);
#pragma unroll
    for (int nt = 0; nt < 4; nt++) {
#pragma unroll
      for (int kk = 0; kk < 4; kk++)
        acc[nt] = wmma_bf(a[kk],
                          load_b(Xh + (kh * 66 + nt * 16 + kw) * LDK + kk * 32, LDK),
                          acc[nt]);
    }
  }
#pragma unroll
  for (int nt = 0; nt < 4; nt++) {
    int n = nt * 16 + r;
    *(uint4*)(Tt + n * LDK + m0 + hg * 8) = pack8_relu(acc[nt]);
  }
  __syncthreads();

  if (wv == 0) {  // tiny 16x128 GEMM (rows 9..15 of wd2b are zero)
    v16bf a2[4];
#pragma unroll
    for (int kk = 0; kk < 4; kk++) a2[kk] = load_a(wd2b + kk * 32, 128);
    unsigned short* ddb = dd + (((size_t)b * HWSZ + h * WW + w0) << 4);
#pragma unroll
    for (int nt = 0; nt < 4; nt++) {
      v8f ac = vzero8();
#pragma unroll
      for (int kk = 0; kk < 4; kk++)
        ac = wmma_bf(a2[kk], load_b(Tt + (nt * 16) * LDK + kk * 32, LDK), ac);
      int n = nt * 16 + r;
      *(uint4*)(ddb + n * 16 + hg * 8) = pack8(ac);
    }
  }
}

// ---------------------------------------------------------------------------
// Kernel 3: out = Wpaka * V, V[k*128+c][n] = xpad * (1 + tanh(dd_k + cc_c)).
// K = 1152 as 9 taps x 128 channels; gate = 2*sigmoid(2s) via v_exp+v_rcp.
// ---------------------------------------------------------------------------
__global__ __launch_bounds__(256) void k_final(
    const unsigned short* __restrict__ xb16, const unsigned short* __restrict__ wpb,
    const unsigned short* __restrict__ cc, const unsigned short* __restrict__ dd,
    float* __restrict__ out) {
  extern __shared__ unsigned short sm3[];
  unsigned short* Xh  = sm3;                 // [3][66][LDK] zero-pad halo
  unsigned short* Vt  = sm3 + 3 * 66 * LDK;  // [64][LDK] gated B operand
  unsigned short* ccs = Vt + 64 * LDK;       // [64][128]
  unsigned short* dds = ccs + 64 * 128;      // [64][16]
  const int tid = threadIdx.x;
  const int b = blockIdx.x >> 8;
  const int rem = blockIdx.x & 255;
  const int h = rem >> 1, w0 = (rem & 1) << 6;
  const unsigned short* xb = xb16 + ((size_t)b * HWSZ << 7);

  // halo: async copy in-bounds rows, ds_store zeros for padding
  const uint4 z4 = make_uint4(0u, 0u, 0u, 0u);
  for (int idx = tid; idx < 3 * 66 * 16; idx += 256) {
    int c8 = (idx & 15) << 3, t = idx >> 4;
    int col = t % 66, rr = t / 66;
    int wi = w0 - 1 + col, hi = h - 1 + rr;
    if (wi >= 0 && wi < WW && hi >= 0 && hi < HH)
      async_cp16(Xh + (rr * 66 + col) * LDK + c8, xb + ((hi * WW + wi) << 7) + c8);
    else
      *(uint4*)(Xh + (rr * 66 + col) * LDK + c8) = z4;
  }
  const unsigned short* ccg = cc + (((size_t)b * HWSZ + h * WW + w0) << 7);
  for (int idx = tid; idx < 64 * 16; idx += 256)
    async_cp16(ccs + idx * 8, ccg + idx * 8);
  const unsigned short* ddg = dd + (((size_t)b * HWSZ + h * WW + w0) << 4);
  for (int idx = tid; idx < 128; idx += 256)
    async_cp16(dds + idx * 8, ddg + idx * 8);
  async_wait0();
  __syncthreads();

  const int wv = tid >> 5, m0 = wv << 4;
  const int lane = tid & 31, r = lane & 15, hg = lane >> 4;

  v8f acc[4];
#pragma unroll
  for (int nt = 0; nt < 4; nt++) acc[nt] = vzero8();

  for (int kp = 0; kp < 9; kp++) {
    int kh = kp / 3, kw = kp - kh * 3;
    // Build gated operand V for this tap.
    for (int idx = tid; idx < 64 * 128; idx += 256) {
      int c = idx & 127, p = idx >> 7;
      float s = bf2f(dds[p * 16 + kp]) + bf2f(ccs[p * 128 + c]);
      // 1 + tanh(s) = 2 / (1 + exp(-2s))
      float g = 2.f * __builtin_amdgcn_rcpf(
                          1.f + __builtin_amdgcn_exp2f(-2.8853900817779268f * s));
      float xv = bf2f(Xh[(kh * 66 + p + kw) * LDK + c]);
      Vt[p * LDK + c] = f2bf(xv * g);
    }
    __syncthreads();
    v16bf a[4];
#pragma unroll
    for (int kk = 0; kk < 4; kk++)
      a[kk] = load_a(wpb + m0 * 1152 + kp * 128 + kk * 32, 1152);
#pragma unroll
    for (int nt = 0; nt < 4; nt++) {
#pragma unroll
      for (int kk = 0; kk < 4; kk++)
        acc[nt] = wmma_bf(a[kk], load_b(Vt + (nt * 16) * LDK + kk * 32, LDK),
                          acc[nt]);
    }
    __syncthreads();
  }

  float* ob = out + (size_t)b * CCH * HWSZ + h * WW + w0;
#pragma unroll
  for (int nt = 0; nt < 4; nt++) {
#pragma unroll
    for (int j = 0; j < 8; j++) {
      int o = m0 + hg * 8 + j;
      ob[(size_t)o * HWSZ + nt * 16 + r] = acc[nt][j];
    }
  }
}

// ---------------------------------------------------------------------------
extern "C" void kernel_launch(void* const* d_in, const int* in_sizes, int n_in,
                              void* d_out, int out_size, void* d_ws, size_t ws_size,
                              hipStream_t stream) {
  const float* x     = (const float*)d_in[0];
  const float* wpaka = (const float*)d_in[1];
  const float* wc1   = (const float*)d_in[2];
  const float* wc2   = (const float*)d_in[3];
  const float* wd1   = (const float*)d_in[4];
  const float* wd2   = (const float*)d_in[5];
  float* out = (float*)d_out;

  char* ws = (char*)d_ws;
  unsigned short* wc1b = (unsigned short*)(ws);                  //    32768 B
  unsigned short* wc2b = (unsigned short*)(ws + 32768);          //    32768 B
  unsigned short* wd1b = (unsigned short*)(ws + 65536);          //   294912 B
  unsigned short* wd2b = (unsigned short*)(ws + 360448);         //     4096 B
  unsigned short* wpb  = (unsigned short*)(ws + 364544);         //   294912 B
  unsigned short* cc   = (unsigned short*)(ws + 659456);         // 16777216 B
  unsigned short* dd   = (unsigned short*)(ws + 17436672);       //  2097152 B
  unsigned short* xb16 = (unsigned short*)(ws + 19533824);       // 16777216 B
  // total: 36311040 B

  const int smem2 = (3 * 66 * LDK + 64 * LDK) * 2;                         // 71264
  const int smem3 = (3 * 66 * LDK + 64 * LDK + 64 * 128 + 64 * 16) * 2;    // 89696
  (void)hipFuncSetAttribute((const void*)k_convd,
                            hipFuncAttributeMaxDynamicSharedMemorySize, smem2);
  (void)hipFuncSetAttribute((const void*)k_final,
                            hipFuncAttributeMaxDynamicSharedMemorySize, smem3);

  k_prep_w<<<576, 256, 0, stream>>>(wc1, wc2, wd1, wd2, wpaka,
                                    wc1b, wc2b, wd1b, wd2b, wpb);
  k_prep_x<<<(BATCH * HWSZ * CCH) / 256, 256, 0, stream>>>(x, xb16);
  k_convc<<<BATCH * (HWSZ / 64), 256, 0, stream>>>(xb16, wc1b, wc2b, cc);
  k_convd<<<BATCH * HH * (WW / 64), 256, smem2, stream>>>(xb16, wd1b, wd2b, dd);
  k_final<<<BATCH * HH * (WW / 64), 256, smem3, stream>>>(xb16, wpb, cc, dd, out);
}
